// MTST_backbone_65137474011823
// MI455X (gfx1250) — matrix operand order repaired and checked by reading, not compiled, not run on hardware
//
#include <hip/hip_runtime.h>
#include <hip/hip_bf16.h>
#include <math.h>
#include <stdint.h>

// ---------------- problem constants ----------------
static constexpr int Lc = 3;
static constexpr int Bc = 256;
static constexpr int Qc = 128;
static constexpr int Pc = 16;
static constexpr int Dc = 256;
static constexpr int Hc = 16;
static constexpr int DKc = 16;
static constexpr int Fc = 512;
static constexpr float SCALEc = 0.25f;      // 16^-0.5
static constexpr float EPSc = 1e-5f;
static constexpr int Mrows = Bc * Qc;       // 32768

typedef __bf16 v16bf __attribute__((ext_vector_type(16)));
typedef float  v8f   __attribute__((ext_vector_type(8)));

#define USE_ASYNC_LDS 1

__device__ __forceinline__ __bf16 f2bf(float f) { return (__bf16)f; }

__device__ __forceinline__ float gelu_exact(float x) {
    return 0.5f * x * (1.0f + erff(x * 0.70710678118654752f));
}

// A-fragment K position for 16-bit A 16x32 (ISA 7.12.2): VGPR vg, lane half
__device__ __forceinline__ int a_kpos(int vg, int half) {
    return (vg < 4) ? (half * 8 + 2 * vg) : (16 + half * 8 + 2 * (vg - 4));
}

// 16-byte global -> LDS copy. Async (ASYNCcnt) on gfx1250, sync fallback otherwise.
__device__ __forceinline__ void copy16_g2l(const __bf16* g, __bf16* l) {
#if USE_ASYNC_LDS
    asm volatile("global_load_async_to_lds_b128 %0, %1, off"
                 :: "v"((unsigned)(uintptr_t)l), "v"(g)
                 : "memory");
#else
    *(uint4*)l = *(const uint4*)g;
#endif
}

template <int N>
__device__ __forceinline__ void wait_async() {
#if USE_ASYNC_LDS
    asm volatile("s_wait_asynccnt %0" :: "n"(N));
#endif
}

// ---------------- f32 -> bf16 conversion ----------------
__global__ __launch_bounds__(256) void cvt_kernel(
    const float* __restrict__ in, __bf16* __restrict__ out, int n)
{
    int i = blockIdx.x * 256 + threadIdx.x;
    if (i < n) out[i] = f2bf(in[i]);
}

// ---------------- embed: z = x @ W_P + b_P + W_pos (f32 + bf16 shadow) ----------------
__global__ __launch_bounds__(256) void embed_kernel(
    const float* __restrict__ x, const float* __restrict__ W_P,
    const float* __restrict__ b_P, const float* __restrict__ W_pos,
    float* __restrict__ z, __bf16* __restrict__ zb)
{
    int row = blockIdx.x;                 // b*Q + q
    int q = row & (Qc - 1);
    int d = threadIdx.x;
    __shared__ float xs[Pc];
    if (threadIdx.x < Pc) xs[threadIdx.x] = x[(size_t)row * Pc + threadIdx.x];
    __syncthreads();
    float acc = b_P[d] + W_pos[(size_t)q * Dc + d];
#pragma unroll
    for (int p = 0; p < Pc; ++p) acc += xs[p] * W_P[p * Dc + d];
    z[(size_t)row * Dc + d] = acc;
    zb[(size_t)row * Dc + d] = f2bf(acc);
}

// ---------------- WMMA bf16 GEMM: C = act(A@W + bias) ----------------
// A: MxK bf16 row-major, W: KxN bf16 row-major. Block 64x128, BK=32.
// 8 waves (2x4), each wave 32x32 (2x2 frags). Double-buffered async LDS staging.
static constexpr int BM = 64, BN = 128, BK = 32;
static constexpr int APITCH = BK + 8;    // 40 bf16 = 80 B rows (16B-aligned chunks)
static constexpr int BPITCH = BN + 8;    // 136 bf16 = 272 B rows

__global__ __launch_bounds__(256) void gemm_bf16_kernel(
    const __bf16* __restrict__ A, const __bf16* __restrict__ W,
    const float* __restrict__ bias, void* __restrict__ Cout,
    int M, int N, int K, int act, int store_bf16)
{
    __shared__ __bf16 As[2][BM * APITCH];
    __shared__ __bf16 Bs[2][BK * BPITCH];

    const int tid  = threadIdx.x;
    const int wave = tid >> 5, lane = tid & 31;
    const int half = lane >> 4, r = lane & 15;
    const int wrow = wave >> 2, wcol = wave & 3;
    const int bm0 = blockIdx.y * BM, bn0 = blockIdx.x * BN;

    // per-stage staging: 3 async (b128) instructions per wave
    auto issue_stage = [&](int buf, int k0) {
        {   // A tile: 64 rows x 4 chunks = 256 chunks, 1/thread
            int m = tid >> 2, c = tid & 3;
            copy16_g2l(A + (size_t)(bm0 + m) * K + k0 + c * 8,
                       &As[buf][m * APITCH + c * 8]);
        }
        {   // B tile: 32 rows x 16 chunks = 512 chunks, 2/thread
            int kk = tid >> 4, c = tid & 15;
            copy16_g2l(W + (size_t)(k0 + kk) * N + bn0 + c * 8,
                       &Bs[buf][kk * BPITCH + c * 8]);
            int t2 = tid + 256;
            int kk2 = t2 >> 4, c2 = t2 & 15;
            copy16_g2l(W + (size_t)(k0 + kk2) * N + bn0 + c2 * 8,
                       &Bs[buf][kk2 * BPITCH + c2 * 8]);
        }
    };

    v8f acc[2][2] = {};
    const int nk = K / BK;
    issue_stage(0, 0);

    for (int i = 0; i < nk; ++i) {
        int cur = i & 1;
        if (i + 1 < nk) {
            issue_stage(cur ^ 1, (i + 1) * BK);
            wait_async<3>();        // stage i complete (in-order), i+1 in flight
        } else {
            wait_async<0>();
        }
        __syncthreads();

        // A fragments: lane->M, vector->K (interleaved pattern)
        v16bf af[2];
#pragma unroll
        for (int mi = 0; mi < 2; ++mi) {
            int arow = wrow * 32 + mi * 16 + r;
#pragma unroll
            for (int vg = 0; vg < 8; ++vg) {
                int ks = a_kpos(vg, half);
                af[mi][2 * vg]     = As[cur][arow * APITCH + ks];
                af[mi][2 * vg + 1] = As[cur][arow * APITCH + ks + 1];
            }
        }
        // B fragments: lane->N (r), vector->K (same interleaved pattern as A)
        v16bf bfr[2];
        int krow = half * 16 + r;
#pragma unroll
        for (int ni = 0; ni < 2; ++ni)
#pragma unroll
            for (int j = 0; j < 16; ++j)
                bfr[ni][j] = Bs[cur][(((j & 8) << 1) + half * 8 + (j & 7)) * BPITCH + wcol * 32 + ni * 16 + r];

#pragma unroll
        for (int mi = 0; mi < 2; ++mi)
#pragma unroll
            for (int ni = 0; ni < 2; ++ni)
                acc[mi][ni] = __builtin_amdgcn_wmma_f32_16x16x32_bf16(
                    false, af[mi], false, bfr[ni], (short)0, acc[mi][ni], false, false);
        __syncthreads();
    }

    // epilogue: C/D layout lane->N (r), VGPR i -> M = i + 8*half
#pragma unroll
    for (int mi = 0; mi < 2; ++mi)
#pragma unroll
        for (int ni = 0; ni < 2; ++ni) {
            int n = bn0 + wcol * 32 + ni * 16 + r;
            float bv = bias ? bias[n] : 0.0f;
#pragma unroll
            for (int i = 0; i < 8; ++i) {
                int m = bm0 + wrow * 32 + mi * 16 + half * 8 + i;
                float v = acc[mi][ni][i] + bv;
                if (act == 1) v = gelu_exact(v);
                if (store_bf16) ((__bf16*)Cout)[(size_t)m * N + n] = f2bf(v);
                else            ((float*)Cout)[(size_t)m * N + n] = v;
            }
        }
}

// ---------------- fused attention per (b,h) ----------------
// S = q k^T * SCALE + prev + bias ; scores <- S (in-place accumulate, NT) ;
// P = softmax_row(S) ; O = P @ v   (Q=128, DK=16 padded to K=32)
__global__ __launch_bounds__(128) void attn_kernel(
    const __bf16* __restrict__ qp, const __bf16* __restrict__ kp,
    const __bf16* __restrict__ vp, const float* __restrict__ attn_bias,
    float* __restrict__ scores, __bf16* __restrict__ oatt, int useprev)
{
    const int bh = blockIdx.x;
    const int b = bh / Hc, h = bh % Hc;
    const int tid = threadIdx.x;
    const int w = tid >> 5, lane = tid & 31;
    const int half = lane >> 4, r = lane & 15;

    __shared__ __bf16 qs[Qc][BK + 2];        // q rows, K padded 16->32 (zeros)
    __shared__ __bf16 kst[BK][Qc + 4];       // k^T: [d(pad32)][j]
    __shared__ __bf16 vs[Qc][DKc + 2];       // v rows: [j][d]
    __shared__ float  sS[Qc][Qc + 4];        // f32 scores
    __shared__ __bf16 sP[Qc][Qc + 4];        // bf16 probabilities

    const __bf16 bz = (__bf16)0.0f;
    for (int i = tid; i < Qc * BK; i += 128) {
        int j = i >> 5, d = i & 31;
        __bf16 qv = (d < DKc) ? qp[((size_t)(b * Qc + j)) * Dc + h * DKc + d] : bz;
        __bf16 kv = (d < DKc) ? kp[((size_t)(b * Qc + j)) * Dc + h * DKc + d] : bz;
        qs[j][d] = qv;
        kst[d][j] = kv;
    }
    for (int i = tid; i < Qc * DKc; i += 128) {
        int j = i >> 4, d = i & 15;
        vs[j][d] = vp[((size_t)(b * Qc + j)) * Dc + h * DKc + d];
    }
    __syncthreads();

    // ---- S = q k^T : wave w owns rows [32w, 32w+32), all 128 cols ----
    v16bf aq[2];
#pragma unroll
    for (int mi = 0; mi < 2; ++mi) {
        int arow = w * 32 + mi * 16 + r;
#pragma unroll
        for (int vg = 0; vg < 8; ++vg) {
            int ks = a_kpos(vg, half);
            aq[mi][2 * vg]     = qs[arow][ks];
            aq[mi][2 * vg + 1] = qs[arow][ks + 1];
        }
    }
#pragma unroll
    for (int ct = 0; ct < 8; ++ct) {
        v16bf bk;
        int krow = half * 16 + r;
#pragma unroll
        for (int j = 0; j < 16; ++j) bk[j] = kst[((j & 8) << 1) + half * 8 + (j & 7)][ct * 16 + r];
#pragma unroll
        for (int mi = 0; mi < 2; ++mi) {
            v8f s = {};
            s = __builtin_amdgcn_wmma_f32_16x16x32_bf16(
                false, aq[mi], false, bk, (short)0, s, false, false);
#pragma unroll
            for (int i = 0; i < 8; ++i) {
                int qi = w * 32 + mi * 16 + half * 8 + i;
                int qj = ct * 16 + r;
                size_t sidx = ((size_t)bh * Qc + qi) * Qc + qj;
                float val = s[i] * SCALEc + attn_bias[qi * Qc + qj];
                if (useprev) val += __builtin_nontemporal_load(&scores[sidx]);
                __builtin_nontemporal_store(val, &scores[sidx]); // carried to next layer
                sS[qi][qj] = val;
            }
        }
    }
    __syncthreads();

    // ---- row softmax: one thread per row ----
    {
        int row = tid;
        float mx = -3.0e38f;
        for (int j = 0; j < Qc; ++j) mx = fmaxf(mx, sS[row][j]);
        float sum = 0.0f;
        for (int j = 0; j < Qc; ++j) sum += __expf(sS[row][j] - mx);
        float inv = 1.0f / sum;
        for (int j = 0; j < Qc; ++j)
            sP[row][j] = f2bf(__expf(sS[row][j] - mx) * inv);
    }
    __syncthreads();

    // ---- O = P @ v : K loop over 128 in steps of 32, N = 16 ----
    v8f oacc[2] = {};
    for (int k0 = 0; k0 < Qc; k0 += BK) {
        v16bf bv;
        int krow = k0 + half * 16 + r;
#pragma unroll
        for (int j = 0; j < 16; ++j) bv[j] = vs[k0 + ((j & 8) << 1) + half * 8 + (j & 7)][r];
#pragma unroll
        for (int mi = 0; mi < 2; ++mi) {
            v16bf ap;
            int arow = w * 32 + mi * 16 + r;
#pragma unroll
            for (int vg = 0; vg < 8; ++vg) {
                int ks = a_kpos(vg, half);
                ap[2 * vg]     = sP[arow][k0 + ks];
                ap[2 * vg + 1] = sP[arow][k0 + ks + 1];
            }
            oacc[mi] = __builtin_amdgcn_wmma_f32_16x16x32_bf16(
                false, ap, false, bv, (short)0, oacc[mi], false, false);
        }
    }
#pragma unroll
    for (int mi = 0; mi < 2; ++mi)
#pragma unroll
        for (int i = 0; i < 8; ++i) {
            int qi = w * 32 + mi * 16 + half * 8 + i;
            oatt[((size_t)(b * Qc + qi)) * Dc + h * DKc + r] = f2bf(oacc[mi][i]);
        }
}

// ---------------- residual + LayerNorm: out = LN(z + o), optional bf16 shadow ----------------
__global__ __launch_bounds__(256) void residual_ln_kernel(
    const float* __restrict__ z, const float* __restrict__ o,
    const float* __restrict__ g, const float* __restrict__ bb,
    float* __restrict__ out, __bf16* __restrict__ outb)
{
    int row = blockIdx.x, d = threadIdx.x;
    float v = z[(size_t)row * Dc + d] + o[(size_t)row * Dc + d];
    __shared__ float red[Dc];
    red[d] = v;
    __syncthreads();
#pragma unroll
    for (int s = Dc / 2; s > 0; s >>= 1) {
        if (d < s) red[d] += red[d + s];
        __syncthreads();
    }
    float mean = red[0] * (1.0f / Dc);
    __syncthreads();
    float dv = v - mean;
    red[d] = dv * dv;
    __syncthreads();
#pragma unroll
    for (int s = Dc / 2; s > 0; s >>= 1) {
        if (d < s) red[d] += red[d + s];
        __syncthreads();
    }
    float var = red[0] * (1.0f / Dc);
    float res = dv * rsqrtf(var + EPSc) * g[d] + bb[d];
    out[(size_t)row * Dc + d] = res;
    if (outb) outb[(size_t)row * Dc + d] = f2bf(res);
}

// ---------------- host orchestration ----------------
extern "C" void kernel_launch(void* const* d_in, const int* in_sizes, int n_in,
                              void* d_out, int out_size, void* d_ws, size_t ws_size,
                              hipStream_t stream) {
    const float* x         = (const float*)d_in[0];
    const float* attn_bias = (const float*)d_in[1];
    const float* W_P       = (const float*)d_in[2];
    const float* b_P       = (const float*)d_in[3];
    const float* W_pos     = (const float*)d_in[4];
    const float* Wq        = (const float*)d_in[5];
    const float* bq        = (const float*)d_in[6];
    const float* Wk        = (const float*)d_in[7];
    const float* bk        = (const float*)d_in[8];
    const float* Wv        = (const float*)d_in[9];
    const float* bv        = (const float*)d_in[10];
    const float* Wo        = (const float*)d_in[11];
    const float* bo        = (const float*)d_in[12];
    const float* ln1_g     = (const float*)d_in[13];
    const float* ln1_b     = (const float*)d_in[14];
    const float* W1        = (const float*)d_in[15];
    const float* b1        = (const float*)d_in[16];
    const float* W2        = (const float*)d_in[17];
    const float* b2        = (const float*)d_in[18];
    const float* ln2_g     = (const float*)d_in[19];
    const float* ln2_b     = (const float*)d_in[20];
    float* out = (float*)d_out;

    const size_t M = (size_t)Mrows;
    char* p = (char*)d_ws;
    auto alloc = [&](size_t bytes) {
        char* ret = p;
        p += (bytes + 255) & ~(size_t)255;
        return ret;
    };
    float*  z      = (float*)alloc(M * Dc * 4);
    float*  proj   = (float*)alloc(M * Dc * 4);
    float*  scores = (float*)alloc((size_t)Bc * Hc * Qc * Qc * 4);
    __bf16* zb     = (__bf16*)alloc(M * Dc * 2);
    __bf16* qb     = (__bf16*)alloc(M * Dc * 2);
    __bf16* kb     = (__bf16*)alloc(M * Dc * 2);
    __bf16* vb     = (__bf16*)alloc(M * Dc * 2);
    __bf16* ob     = (__bf16*)alloc(M * Dc * 2);
    __bf16* fhb    = (__bf16*)alloc(M * Fc * 2);
    __bf16* wqb    = (__bf16*)alloc((size_t)Lc * Dc * Dc * 2);
    __bf16* wkb    = (__bf16*)alloc((size_t)Lc * Dc * Dc * 2);
    __bf16* wvb    = (__bf16*)alloc((size_t)Lc * Dc * Dc * 2);
    __bf16* wob    = (__bf16*)alloc((size_t)Lc * Dc * Dc * 2);
    __bf16* w1b    = (__bf16*)alloc((size_t)Lc * Dc * Fc * 2);
    __bf16* w2b    = (__bf16*)alloc((size_t)Lc * Fc * Dc * 2);

    dim3 blk256(256), blk128(128);
    dim3 g_rows(Mrows);
    dim3 g_gemmD(Dc / BN, Mrows / BM);       // N=256
    dim3 g_gemmF(Fc / BN, Mrows / BM);       // N=512
    dim3 g_attn(Bc * Hc);

    // one-time weight conversion to bf16
    const int wDD = Lc * Dc * Dc, wDF = Lc * Dc * Fc;
    cvt_kernel<<<(wDD + 255) / 256, blk256, 0, stream>>>(Wq, wqb, wDD);
    cvt_kernel<<<(wDD + 255) / 256, blk256, 0, stream>>>(Wk, wkb, wDD);
    cvt_kernel<<<(wDD + 255) / 256, blk256, 0, stream>>>(Wv, wvb, wDD);
    cvt_kernel<<<(wDD + 255) / 256, blk256, 0, stream>>>(Wo, wob, wDD);
    cvt_kernel<<<(wDF + 255) / 256, blk256, 0, stream>>>(W1, w1b, wDF);
    cvt_kernel<<<(wDF + 255) / 256, blk256, 0, stream>>>(W2, w2b, wDF);

    embed_kernel<<<g_rows, blk256, 0, stream>>>(x, W_P, b_P, W_pos, z, zb);

    for (int l = 0; l < Lc; ++l) {
        const __bf16* wq = wqb + (size_t)l * Dc * Dc;
        const __bf16* wk = wkb + (size_t)l * Dc * Dc;
        const __bf16* wv = wvb + (size_t)l * Dc * Dc;
        const __bf16* wo = wob + (size_t)l * Dc * Dc;
        const __bf16* w1 = w1b + (size_t)l * Dc * Fc;
        const __bf16* w2 = w2b + (size_t)l * Fc * Dc;

        gemm_bf16_kernel<<<g_gemmD, blk256, 0, stream>>>(zb, wq, bq + l * Dc, qb, Mrows, Dc, Dc, 0, 1);
        gemm_bf16_kernel<<<g_gemmD, blk256, 0, stream>>>(zb, wk, bk + l * Dc, kb, Mrows, Dc, Dc, 0, 1);
        gemm_bf16_kernel<<<g_gemmD, blk256, 0, stream>>>(zb, wv, bv + l * Dc, vb, Mrows, Dc, Dc, 0, 1);

        attn_kernel<<<g_attn, blk128, 0, stream>>>(qb, kb, vb, attn_bias,
                                                   scores, ob, l > 0 ? 1 : 0);

        gemm_bf16_kernel<<<g_gemmD, blk256, 0, stream>>>(ob, wo, bo + l * Dc, proj, Mrows, Dc, Dc, 0, 0);
        residual_ln_kernel<<<g_rows, blk256, 0, stream>>>(z, proj, ln1_g + l * Dc, ln1_b + l * Dc, z, zb);

        gemm_bf16_kernel<<<g_gemmF, blk256, 0, stream>>>(zb, w1, b1 + l * Fc, fhb, Mrows, Fc, Dc, 1, 1);
        gemm_bf16_kernel<<<g_gemmD, blk256, 0, stream>>>(fhb, w2, b2 + l * Dc, proj, Mrows, Dc, Fc, 0, 0);
        residual_ln_kernel<<<g_rows, blk256, 0, stream>>>(z, proj, ln2_g + l * Dc, ln2_b + l * Dc,
                                                          (l == Lc - 1) ? out : z,
                                                          (l == Lc - 1) ? (__bf16*)nullptr : zb);
    }
}
